// SA2_45612552683709
// MI455X (gfx1250) — matrix operand order repaired and checked by reading, not compiled, not run on hardware
//
#include <hip/hip_runtime.h>
#include <hip/hip_fp16.h>

// ---------------------------------------------------------------------------
// Problem constants (match reference)
// ---------------------------------------------------------------------------
typedef _Float16 h16;
typedef __attribute__((ext_vector_type(16))) _Float16 v16h;
typedef __attribute__((ext_vector_type(8)))  _Float16 v8h;
typedef __attribute__((ext_vector_type(8)))  float    v8f;

constexpr int BB   = 4;
constexpr int T    = 16;
constexpr int H    = 32;
constexpr int W    = 32;
constexpr int C    = 512;
constexpr int G    = 16;
constexpr int CG   = C / G;          // 32 channels per group
constexpr int HW   = H * W;          // 1024
constexpr int THW  = T * HW;         // 16384
constexpr int NTOK = BB * THW;       // 65536 tokens
constexpr float EPS = 1e-6f;

// ---------------------------------------------------------------------------
// CDNA5 async global->LDS copy (16 bytes per lane), tracked by ASYNCcnt.
// GV addressing: 64-bit global address in a VGPR pair, per-lane LDS byte
// address in VDST.  Flat pointers to LDS carry the LDS offset in addr[31:0].
// ---------------------------------------------------------------------------
__device__ __forceinline__ void async_copy_b128(h16* lds, const h16* gptr) {
  unsigned laddr = (unsigned)(unsigned long long)lds;
  asm volatile("global_load_async_to_lds_b128 %0, %1, off"
               :: "v"(laddr), "v"((unsigned long long)gptr)
               : "memory");
}

__device__ __forceinline__ void wait_asynccnt0() {
  asm volatile("s_wait_asynccnt 0x0" ::: "memory");
}

// ---------------------------------------------------------------------------
// Kernel 1: GroupNorm statistics. One block per (b, g); reduce 16384 tokens
// x 32 channels. Produces mean[b*G+g], rstd[b*G+g].
// ---------------------------------------------------------------------------
__global__ void gn_stats(const float* __restrict__ x,
                         float* __restrict__ mean, float* __restrict__ rstd) {
  const int bg = blockIdx.x;            // 0..63
  const int b  = bg / G, g = bg % G;
  const float* base = x + (size_t)b * THW * C + g * CG;
  float s = 0.f, s2 = 0.f;
  for (int tok = threadIdx.x; tok < THW; tok += blockDim.x) {
    const float4* p = (const float4*)(base + (size_t)tok * C);
#pragma unroll
    for (int i = 0; i < CG / 4; ++i) {
      float4 v = p[i];
      s  += v.x + v.y + v.z + v.w;
      s2 += v.x * v.x + v.y * v.y + v.z * v.z + v.w * v.w;
    }
  }
  __shared__ float ss[256], ss2[256];
  ss[threadIdx.x] = s; ss2[threadIdx.x] = s2;
  __syncthreads();
  for (int off = 128; off > 0; off >>= 1) {
    if ((int)threadIdx.x < off) {
      ss[threadIdx.x]  += ss[threadIdx.x + off];
      ss2[threadIdx.x] += ss2[threadIdx.x + off];
    }
    __syncthreads();
  }
  if (threadIdx.x == 0) {
    const float n = (float)THW * (float)CG;
    float m = ss[0] / n;
    float v = ss2[0] / n - m * m;
    mean[bg] = m;
    rstd[bg] = rsqrtf(v + EPS);
  }
}

// ---------------------------------------------------------------------------
// Kernel 2: normalize + affine + cast f32 -> f16, token-major [NTOK][C].
// ---------------------------------------------------------------------------
__global__ void gn_apply(const float* __restrict__ x,
                         const float* __restrict__ gamma,
                         const float* __restrict__ beta,
                         const float* __restrict__ mean,
                         const float* __restrict__ rstd,
                         h16* __restrict__ hn) {
  size_t i4 = ((size_t)blockIdx.x * blockDim.x + threadIdx.x) * 4;
  if (i4 >= (size_t)NTOK * C) return;
  const int    c0  = (int)(i4 % C);
  const size_t tok = i4 / C;
  const int    b   = (int)(tok / THW);
  const int    g   = c0 / CG;
  const float  m   = mean[b * G + g];
  const float  rs  = rstd[b * G + g];
  float4 xv = *(const float4*)(x + i4);
  float4 gv = *(const float4*)(gamma + c0);
  float4 bv = *(const float4*)(beta + c0);
  hn[i4 + 0] = (h16)((xv.x - m) * rs * gv.x + bv.x);
  hn[i4 + 1] = (h16)((xv.y - m) * rs * gv.y + bv.y);
  hn[i4 + 2] = (h16)((xv.z - m) * rs * gv.z + bv.z);
  hn[i4 + 3] = (h16)((xv.w - m) * rs * gv.w + bv.w);
}

// ---------------------------------------------------------------------------
// Kernel 3: f32 -> f16 weight conversion.
// ---------------------------------------------------------------------------
__global__ void f32_to_f16(const float* __restrict__ in, h16* __restrict__ out,
                           int n) {
  int i = blockIdx.x * blockDim.x + threadIdx.x;
  if (i < n) out[i] = (h16)in[i];
}

// ---------------------------------------------------------------------------
// Stage one 128x32 A-tile + 64x32 B^T-tile into LDS via async copies.
// 256 threads: 2 A chunks + 1 B chunk of 16B each  => 3 async ops / wave.
// ---------------------------------------------------------------------------
__device__ __forceinline__ void stage_tiles(const h16* __restrict__ A,
                                            const h16* __restrict__ Bw,
                                            h16* sa, h16* sb,
                                            int m0, int n0, int k0, int tid) {
#pragma unroll
  for (int it = 0; it < 2; ++it) {
    int chunk = tid + it * 256;
    int row = chunk >> 2, kc = (chunk & 3) * 8;
    async_copy_b128(sa + row * 32 + kc,
                    A + (size_t)(m0 + row) * C + k0 + kc);
  }
  {
    int row = tid >> 2, kc = (tid & 3) * 8;
    async_copy_b128(sb + row * 32 + kc,
                    Bw + (size_t)(n0 + row) * C + k0 + kc);
  }
}

// ---------------------------------------------------------------------------
// Kernel 4: tiled WMMA GEMM with double-buffered async LDS staging.
//   y[n][o] = sum_c A[n][c] * Wt[o][c] + bias[o]
//   A: [M=65536][K=512] f16 token-major; W: [O=512][C=512] f16 row-major
//   (the LDS "B^T" tile [n][k] is a contiguous slice of W).
// 256 threads = 8 waves as 4(M) x 2(N); block tile 128x64; 2x2 WMMA / wave.
// MODE 0: f16 token-major out.  MODE 1: f32 scattered to (b,c,s).
//
// Wave32 fragment layouts per CDNA5 ISA 7.12.2:
//   A 16x32 f16 : lane holds row M=l%16; lanes<16 K {0..7,16..23},
//                 lanes>=16 K {8..15,24..31}.
//   B 32x16 f16 : lane holds col N=l%16; K as A; this code loads A and B with K 0..15 (lanes<16), 16..31 (lanes>=16).
//   C/D 16x16 f32: VGPR r, lane l -> M = r + (l>=16 ? 8 : 0), N = l%16.
// ---------------------------------------------------------------------------
template <int MODE>
__global__ void gemm_wmma(const h16* __restrict__ A, const h16* __restrict__ Bw,
                          const float* __restrict__ bias,
                          void* __restrict__ outp) {
  __shared__ h16 sA[2][128 * 32];       // 2 x 8 KB
  __shared__ h16 sB[2][64 * 32];        // 2 x 4 KB
  const int tid  = threadIdx.x;
  const int wid  = tid >> 5;
  const int lane = tid & 31;
  const int m0   = blockIdx.x * 128;
  const int n0   = blockIdx.y * 64;
  const int wm   = wid >> 1;            // 0..3
  const int wn   = wid & 1;             // 0..1
  const int l16  = lane & 15;
  const bool hiL = lane >= 16;
  const int ka0  = hiL ? 16 : 0;        // A-frag first 8-half chunk (K offset)
  const int ka1  = hiL ? 24 : 8;        // A-frag second chunk
  const int kb0  = hiL ? 16 : 0;        // B-frag 16 contiguous K values

  v8f acc[2][2];
#pragma unroll
  for (int mt = 0; mt < 2; ++mt)
#pragma unroll
    for (int nt = 0; nt < 2; ++nt) { v8f z = {}; acc[mt][nt] = z; }

  // Prologue: stage k-tile 0 into buffer 0.
  stage_tiles(A, Bw, sA[0], sB[0], m0, n0, 0, tid);

  constexpr int KSTEPS = C / 32;        // 16
  for (int k = 0; k < KSTEPS; ++k) {
    const int cur = k & 1;
    // Own async copies for buf[cur] retired; barrier makes them (and every
    // wave's frag reads of buf[cur^1] from iter k-1) globally visible/done.
    wait_asynccnt0();
    __syncthreads();
    if (k + 1 < KSTEPS)
      stage_tiles(A, Bw, sA[cur ^ 1], sB[cur ^ 1], m0, n0, (k + 1) * 32, tid);

    v16h af[2], bf[2];
#pragma unroll
    for (int mt = 0; mt < 2; ++mt) {
      const h16* rp = sA[cur] + (wm * 32 + mt * 16 + l16) * 32;
      v8h lo = *(const v8h*)(rp + ka0);
      v8h hh = *(const v8h*)(rp + ka1);
      af[mt] = __builtin_shufflevector(lo, hh, 0, 1, 2, 3, 4, 5, 6, 7, 8, 9,
                                       10, 11, 12, 13, 14, 15);
    }
#pragma unroll
    for (int nt = 0; nt < 2; ++nt) {
      const h16* rp = sB[cur] + (wn * 32 + nt * 16 + l16) * 32 + kb0;
      v8h lo = *(const v8h*)(rp);
      v8h hh = *(const v8h*)(rp + 8);
      bf[nt] = __builtin_shufflevector(lo, hh, 0, 1, 2, 3, 4, 5, 6, 7, 8, 9,
                                       10, 11, 12, 13, 14, 15);
    }
#pragma unroll
    for (int mt = 0; mt < 2; ++mt)
#pragma unroll
      for (int nt = 0; nt < 2; ++nt)
        acc[mt][nt] = __builtin_amdgcn_wmma_f32_16x16x32_f16(
            false, af[mt], false, bf[nt], (short)0, acc[mt][nt], false, false);
  }

  // Epilogue: bias + store.
#pragma unroll
  for (int nt = 0; nt < 2; ++nt) {
    const int col = n0 + wn * 32 + nt * 16 + l16;
    const float bv = bias[col];
#pragma unroll
    for (int mt = 0; mt < 2; ++mt) {
#pragma unroll
      for (int r = 0; r < 8; ++r) {
        const int row = m0 + wm * 32 + mt * 16 + r + (hiL ? 8 : 0);
        const float v = acc[mt][nt][r] + bv;
        if (MODE == 0) {
          ((h16*)outp)[(size_t)row * C + col] = (h16)v;
        } else {
          const int b = row / THW, s = row % THW;     // token -> (b, thw)
          ((float*)outp)[((size_t)b * C + col) * THW + s] = v;
        }
      }
    }
  }
}

// ---------------------------------------------------------------------------
// Kernel 5: temporal attention per (b,h,w). One wave per location.
//   scores = q k^T / sqrt(C)   (16x16 via 16 chained WMMAs, K=512)
//   softmax rows               (shfl_xor over 16-lane N groups)
//   o = attn v                 (K=16 zero-padded to 32, 32 WMMAs over N=512)
// Q/K/V/O are f16 token-major [NTOK][C]; for fixed (b,hw) the 16 time rows
// are strided by HW*C.
// ---------------------------------------------------------------------------
__global__ void attn_kernel(const h16* __restrict__ Q, const h16* __restrict__ K,
                            const h16* __restrict__ V, h16* __restrict__ O) {
  __shared__ h16 sAttn[4][16 * 16];
  const int wid  = threadIdx.x >> 5;
  const int lane = threadIdx.x & 31;
  const int loc  = blockIdx.x * 4 + wid;     // 0..4095
  const int b    = loc / HW;
  const int hw   = loc % HW;
  const int l16  = lane & 15;
  const bool hiL = lane >= 16;
  const int ka0  = hiL ? 16 : 0, ka1 = hiL ? 24 : 8, kb0 = hiL ? 16 : 0;
  const size_t tstride = (size_t)HW * C;
  const size_t base    = ((size_t)b * THW + hw) * C;   // t = 0 token

  // ---- scores = q @ k^T -------------------------------------------------
  v8f sc = {};
  const h16* qrow = Q + base + (size_t)l16 * tstride;  // A row  = t_q
  const h16* krow = K + base + (size_t)l16 * tstride;  // B col  = t_k
  for (int k0 = 0; k0 < C; k0 += 32) {
    v8h alo = *(const v8h*)(qrow + k0 + ka0);
    v8h ahi = *(const v8h*)(qrow + k0 + ka1);
    v16h af = __builtin_shufflevector(alo, ahi, 0, 1, 2, 3, 4, 5, 6, 7, 8, 9,
                                      10, 11, 12, 13, 14, 15);
    v8h blo = *(const v8h*)(krow + k0 + kb0);
    v8h bhi = *(const v8h*)(krow + k0 + kb0 + 8);
    v16h bf = __builtin_shufflevector(blo, bhi, 0, 1, 2, 3, 4, 5, 6, 7, 8, 9,
                                      10, 11, 12, 13, 14, 15);
    sc = __builtin_amdgcn_wmma_f32_16x16x32_f16(false, af, false, bf, (short)0,
                                                sc, false, false);
  }

  // ---- softmax over N (the 16 lanes of each half-wave) -------------------
  const float scale = 0.044194173824159216f;   // 512^-0.5
#pragma unroll
  for (int r = 0; r < 8; ++r) {
    float v = sc[r] * scale;
    float m = v;
    m = fmaxf(m, __shfl_xor(m, 1));
    m = fmaxf(m, __shfl_xor(m, 2));
    m = fmaxf(m, __shfl_xor(m, 4));
    m = fmaxf(m, __shfl_xor(m, 8));
    float e = __expf(v - m);
    float s = e;
    s += __shfl_xor(s, 1);
    s += __shfl_xor(s, 2);
    s += __shfl_xor(s, 4);
    s += __shfl_xor(s, 8);
    sc[r] = e / s;
  }

  // ---- stage attn (16x16 f16) in LDS, rebuild as A fragment --------------
  const int Mb = hiL ? 8 : 0;
#pragma unroll
  for (int r = 0; r < 8; ++r)
    sAttn[wid][(r + Mb) * 16 + l16] = (h16)sc[r];
  __syncthreads();

  v8h zero8 = {};
  v8h alo = *(const v8h*)(&sAttn[wid][l16 * 16 + (hiL ? 8 : 0)]);
  // K = 0..15 real attn, K = 16..31 zero padding.
  v16h af = __builtin_shufflevector(alo, zero8, 0, 1, 2, 3, 4, 5, 6, 7, 8, 9,
                                    10, 11, 12, 13, 14, 15);

  // ---- o = attn @ v ------------------------------------------------------
  const h16* vb = V + base;
  for (int c0 = 0; c0 < C; c0 += 16) {
    v16h bf;
#pragma unroll
    for (int t = 0; t < 16; ++t) {
      h16 val = vb[(size_t)((t & 7) + (hiL ? 8 : 0)) * tstride + c0 + l16];
      bf[t] = (t >= 8) ? (h16)0 : val;   // elements 8..15 hold K=16..31 -> zeros
    }
    v8f z = {};
    v8f oc = __builtin_amdgcn_wmma_f32_16x16x32_f16(false, af, false, bf,
                                                    (short)0, z, false, false);
#pragma unroll
    for (int r = 0; r < 8; ++r) {
      const int tq = r + Mb;
      O[base + (size_t)tq * tstride + c0 + l16] = (h16)oc[r];
    }
  }
}

// ---------------------------------------------------------------------------
// Host launch.  Workspace layout (bytes):
//   [0 .. 2MB)       : wq16, wk16, wv16, wp16   (512KB each)
//   [2MB .. 2MB+512) : mean[64], rstd[64]
//   [4MB .. +64MB)   : hn f16 (reused as attention output O)
//   then q, k, v f16 buffers (64MB each)  -> total ~260MB
// ---------------------------------------------------------------------------
extern "C" void kernel_launch(void* const* d_in, const int* in_sizes, int n_in,
                              void* d_out, int out_size, void* d_ws,
                              size_t ws_size, hipStream_t stream) {
  const float* x     = (const float*)d_in[0];
  const float* gamma = (const float*)d_in[1];
  const float* beta  = (const float*)d_in[2];
  const float* wq    = (const float*)d_in[3];
  const float* bq    = (const float*)d_in[4];
  const float* wk    = (const float*)d_in[5];
  const float* bk    = (const float*)d_in[6];
  const float* wv    = (const float*)d_in[7];
  const float* bv    = (const float*)d_in[8];
  const float* wp    = (const float*)d_in[9];
  const float* bp    = (const float*)d_in[10];

  char* ws = (char*)d_ws;
  const size_t WSZ   = (size_t)C * C * sizeof(h16);     // 512 KB
  h16* wq16 = (h16*)(ws + 0 * WSZ);
  h16* wk16 = (h16*)(ws + 1 * WSZ);
  h16* wv16 = (h16*)(ws + 2 * WSZ);
  h16* wp16 = (h16*)(ws + 3 * WSZ);
  float* mean = (float*)(ws + (2ull << 20));
  float* rstd = mean + 64;
  const size_t TOKSZ = (size_t)NTOK * C * sizeof(h16);  // 64 MB
  h16* hn = (h16*)(ws + (4ull << 20));                  // reused as O
  h16* qb = (h16*)((char*)hn + 1 * TOKSZ);
  h16* kb = (h16*)((char*)hn + 2 * TOKSZ);
  h16* vb = (h16*)((char*)hn + 3 * TOKSZ);

  // Weight conversion: 262144 elements each.
  const int nW = C * C;
  f32_to_f16<<<(nW + 255) / 256, 256, 0, stream>>>(wq, wq16, nW);
  f32_to_f16<<<(nW + 255) / 256, 256, 0, stream>>>(wk, wk16, nW);
  f32_to_f16<<<(nW + 255) / 256, 256, 0, stream>>>(wv, wv16, nW);
  f32_to_f16<<<(nW + 255) / 256, 256, 0, stream>>>(wp, wp16, nW);

  // GroupNorm.
  gn_stats<<<BB * G, 256, 0, stream>>>(x, mean, rstd);
  const size_t nApply = ((size_t)NTOK * C) / 4;
  gn_apply<<<(unsigned)((nApply + 255) / 256), 256, 0, stream>>>(
      x, gamma, beta, mean, rstd, hn);

  // QKV projections (f16 out, token-major).
  dim3 gg(NTOK / 128, C / 64);          // (512, 8)
  gemm_wmma<0><<<gg, 256, 0, stream>>>(hn, wq16, bq, qb);
  gemm_wmma<0><<<gg, 256, 0, stream>>>(hn, wk16, bk, kb);
  gemm_wmma<0><<<gg, 256, 0, stream>>>(hn, wv16, bv, vb);

  // Temporal attention; writes O into the (now free) hn buffer.
  attn_kernel<<<(BB * HW) / 4, 128, 0, stream>>>(qb, kb, vb, hn);

  // Output projection: f32 scattered into (b, c, t, h, w).
  gemm_wmma<1><<<gg, 256, 0, stream>>>(hn, wp16, bp, (float*)d_out);
}
